// DHOModelBlack_31387620999407
// MI455X (gfx1250) — matrix operand order repaired and checked by reading, not compiled, not run on hardware
//
#include <hip/hip_runtime.h>

// ---------------------------------------------------------------------------
// DHOModelBlack fused kernel for gfx1250 (MI455X).
//   f_X  = X @ W_fnn^T
//   g_X  = X @ W_gnn^T + b_gnn
//   ICNN forward (60-dim hidden) + analytic per-sample gradient dV
//   out  = f_X - (dV/||dV||^2) * relu(dV.f + a*V - |dV.g|) + g*U
// The two (N x 64)(64 x 64) GEMMs run on v_wmma_f32_16x16x32_f16.
// Transcendentals use raw v_exp_f32 / v_log_f32 / v_rcp_f32, branch-free.
// ---------------------------------------------------------------------------

typedef __attribute__((ext_vector_type(16))) _Float16 v16h;
typedef __attribute__((ext_vector_type(8)))  float    v8f;

#define EPS_C   0.01f
#define ALPHA_C 0.1f

#define LOG2E_C 1.44269504088896340736f
#define LN2_C   0.69314718055994530942f

// branch-free softplus: max(x,0) + ln(1 + exp(-|x|))
//   exp2 arg <= 0  -> no overflow; log2 arg in (1,2] -> no denorm fixups.
__device__ __forceinline__ float sp_f(float x) {
    const float t = __builtin_amdgcn_exp2f(-LOG2E_C * __builtin_fabsf(x));
    return __builtin_fmaxf(x, 0.f) + LN2_C * __builtin_amdgcn_logf(1.f + t);
}
// branch-free sigmoid: 1 / (1 + exp(-x))
__device__ __forceinline__ float sig_f(float x) {
    const float t = __builtin_amdgcn_exp2f(-LOG2E_C * x);
    return __builtin_amdgcn_rcpf(1.f + t);
}

// wave-local LDS ordering (each wave only touches its own LDS region)
#define WAVE_LDS_FENCE() asm volatile("s_wait_dscnt 0x0" ::: "memory")

// ------------------------- kernel 1: h0 = ICNN(Xref) ------------------------
__global__ void k_h0(const float* __restrict__ Xref,
                     const float* __restrict__ W1, const float* __restrict__ b1,
                     const float* __restrict__ W2, const float* __restrict__ b2,
                     const float* __restrict__ W3, const float* __restrict__ b3,
                     const float* __restrict__ Wim2, const float* __restrict__ Wim3,
                     float* __restrict__ h0out) {
    __shared__ float z1[60];
    __shared__ float z2[60];
    const int t  = threadIdx.x;
    const float x0 = Xref[0], x1 = Xref[1];
    if (t < 60) z1[t] = sp_f(W1[t*2]*x0 + W1[t*2+1]*x1 + b1[t]);
    __syncthreads();
    if (t < 60) {
        float a = b2[t] + Wim2[t*2]*x0 + Wim2[t*2+1]*x1;
        for (int k = 0; k < 60; ++k) a += W2[t*60 + k] * z1[k];
        z2[t] = sp_f(a);
    }
    __syncthreads();
    if (t == 0) {
        float a = b3[0] + Wim3[0]*x0 + Wim3[1]*x1;
        for (int k = 0; k < 60; ++k) a += W3[k] * z2[k];
        h0out[0] = sp_f(a);
    }
}

// ------------------------- kernel 2: fused model ----------------------------
// 256 threads = 8 wave32; each wave handles 16 samples (block = 128 samples).
__global__ __launch_bounds__(256) void k_main(
    const float* __restrict__ X,    const float* __restrict__ U,
    const float* __restrict__ Xref, const float* __restrict__ W_fnn,
    const float* __restrict__ W_gnn,const float* __restrict__ b_gnn,
    const float* __restrict__ W1,   const float* __restrict__ b1,
    const float* __restrict__ W2,   const float* __restrict__ b2,
    const float* __restrict__ W3,   const float* __restrict__ b3,
    const float* __restrict__ Wim2, const float* __restrict__ Wim3,
    const float* __restrict__ h0p,  float* __restrict__ out, int N)
{
    __shared__ _Float16 w2h[64*64];        // W2 padded to 64x64, fp16
    __shared__ float    w1s[128];          // W1 padded to 64x2
    __shared__ float    b1s[64], b2s[64], w3s[64];
    __shared__ float    wim2s[128];        // Wim2 padded to 64x2
    __shared__ float    cb[16];            // small constants
    __shared__ float    xb[8][16][2];      // per-wave X tile
    __shared__ _Float16 tp[8][16*64];      // per-wave transpose buffer (t2)
    __shared__ float    hxs[8][16];        // per-wave layer-3 pre-activation
    __shared__ float    g3s[8][16];        // per-wave g3 = sigp*s3
    __shared__ float    dns[8][16][2];     // per-wave reduced gradient

    const int tid = threadIdx.x;

    // ---- stage weights into LDS (block-wide, once) ----
    for (int i = tid; i < 4096; i += 256) {
        const int j = i >> 6, k = i & 63;
        w2h[i] = (_Float16)((j < 60 && k < 60) ? W2[j*60 + k] : 0.f);
    }
    if (tid < 128) {
        w1s[tid]   = (tid < 120) ? W1[tid]   : 0.f;
        wim2s[tid] = (tid < 120) ? Wim2[tid] : 0.f;
    }
    if (tid < 64) {
        b1s[tid] = (tid < 60) ? b1[tid] : 0.f;
        b2s[tid] = (tid < 60) ? b2[tid] : 0.f;
        w3s[tid] = (tid < 60) ? W3[tid] : 0.f;
    }
    if (tid == 0) {
        cb[0] = b3[0];   cb[1] = Wim3[0]; cb[2] = Wim3[1];
        cb[3] = W_fnn[0];cb[4] = W_fnn[1];cb[5] = W_fnn[2];cb[6] = W_fnn[3];
        cb[7] = W_gnn[0];cb[8] = W_gnn[1];cb[9] = W_gnn[2];cb[10]= W_gnn[3];
        cb[11]= b_gnn[0];cb[12]= b_gnn[1];cb[13]= Xref[0]; cb[14]= Xref[1];
        cb[15]= h0p[0];
    }
    __syncthreads();

    const int  wv   = tid >> 5;
    const int  ln   = tid & 31;
    const int  half = ln >> 4;        // 0: lanes 0-15, 1: lanes 16-31
    const int  lm   = ln & 15;
    const long long base = (long long)blockIdx.x * 128 + (long long)wv * 16;

    // ---- stage this wave's 16 X rows ----
    if (half == 0) {
        long long s = base + lm;
        if (s >= N) s = N - 1;
        xb[wv][lm][0] = X[s*2 + 0];
        xb[wv][lm][1] = X[s*2 + 1];
    }
    WAVE_LDS_FENCE();

    // ---- layer 1: build z1 directly in WMMA A-layout (row = lane&15) ----
    // A 16x32 fp16 layout: half i -> K = i + (i<8 ? (hi?8:0) : (hi?16:8))
    v16h A0, A1;
    {
        const float x0 = xb[wv][lm][0], x1 = xb[wv][lm][1];
        #pragma unroll
        for (int i = 0; i < 16; ++i) {
            const int kb = i + ((i < 8) ? (half ? 8 : 0) : (half ? 16 : 8));
            const int k0 = kb, k1 = 32 + kb;
            const float v0 = (k0 < 60) ? sp_f(x0*w1s[k0*2] + x1*w1s[k0*2+1] + b1s[k0]) : 0.f;
            const float v1 = (k1 < 60) ? sp_f(x0*w1s[k1*2] + x1*w1s[k1*2+1] + b1s[k1]) : 0.f;
            A0[i] = (_Float16)v0;
            A1[i] = (_Float16)v1;
        }
    }

    // ---- layer 2 forward GEMM: u2 = z1 @ W2^T  (4 col-tiles x 2 K-chunks) ----
    float a2s[4][8];
    float hxp[8];
    #pragma unroll
    for (int r = 0; r < 8; ++r) hxp[r] = 0.f;

    #pragma unroll
    for (int t = 0; t < 4; ++t) {
        const int col = 16*t + lm;
        v16h B0, B1;
        #pragma unroll
        for (int i = 0; i < 16; ++i) {
            const int kb = i + ((i < 8) ? (half ? 8 : 0) : (half ? 16 : 8));       // B 32x16 fp16 layout
            B0[i] = w2h[col*64 + kb];                 // B[k][n] = W2[col][k]
            B1[i] = w2h[col*64 + 32 + kb];
        }
        v8f acc = {};
        acc = __builtin_amdgcn_wmma_f32_16x16x32_f16(false, A0, false, B0,
                                                     (short)0, acc, false, false);
        acc = __builtin_amdgcn_wmma_f32_16x16x32_f16(false, A1, false, B1,
                                                     (short)0, acc, false, false);
        const float wim2x = wim2s[col*2], wim2y = wim2s[col*2+1];
        const float bb = b2s[col], w3c = w3s[col];
        #pragma unroll
        for (int r = 0; r < 8; ++r) {                 // D layout: row=r(+8), col=lane&15
            const int row = r + (half ? 8 : 0);
            const float rx0 = xb[wv][row][0], rx1 = xb[wv][row][1];
            const float a2  = acc[r] + bb + rx0*wim2x + rx1*wim2y;
            a2s[t][r] = a2;
            hxp[r] += sp_f(a2) * w3c;                 // layer-3 dot partial
        }
    }

    // ---- reduce layer-3 dot across the 16 lanes of each half ----
    #pragma unroll
    for (int r = 0; r < 8; ++r) {
        float v = hxp[r];
        v += __shfl_xor(v, 1, 16);
        v += __shfl_xor(v, 2, 16);
        v += __shfl_xor(v, 4, 16);
        v += __shfl_xor(v, 8, 16);
        hxp[r] = v;
    }
    if (lm == 0) {
        #pragma unroll
        for (int r = 0; r < 8; ++r) hxs[wv][r + (half ? 8 : 0)] = hxp[r];
    }
    WAVE_LDS_FENCE();

    // ---- per-sample scalars: V, sigma', g3 = sigma'*s3 (lanes 0-15) ----
    float Vm = 0.f, g3m = 0.f;
    if (half == 0) {
        const float x0 = xb[wv][lm][0], x1 = xb[wv][lm][1];
        const float a3 = hxs[wv][lm] + cb[0] + x0*cb[1] + x1*cb[2];
        const float hx = sp_f(a3);
        const float h  = hx - cb[15];
        // piecewise huber: branch-free selects
        const float sigma = (h >= 1.f) ? (h - 0.5f) : ((h > 0.f) ? 0.5f*h*h : 0.f);
        const float sigp  = (h >= 1.f) ? 1.f        : ((h > 0.f) ? h        : 0.f);
        const float dx0 = x0 - cb[13], dx1 = x1 - cb[14];
        Vm  = sigma + EPS_C*(dx0*dx0 + dx1*dx1);
        g3m = sigp * sig_f(a3);
        g3s[wv][lm] = g3m;
    }
    WAVE_LDS_FENCE();

    // ---- build t2 = g3 * W3[j] * s2[j], stage through LDS for transpose ----
    float dnx[8], dny[8], g3r[8];
    #pragma unroll
    for (int r = 0; r < 8; ++r) { dnx[r] = 0.f; dny[r] = 0.f; }
    #pragma unroll
    for (int r = 0; r < 8; ++r) g3r[r] = g3s[wv][r + (half ? 8 : 0)];

    #pragma unroll
    for (int t = 0; t < 4; ++t) {
        const int col = 16*t + lm;
        const float w3c = w3s[col];
        const float wim2x = wim2s[col*2], wim2y = wim2s[col*2+1];
        #pragma unroll
        for (int r = 0; r < 8; ++r) {
            const int row = r + (half ? 8 : 0);
            const float t2 = g3r[r] * w3c * sig_f(a2s[t][r]);
            tp[wv][row*64 + col] = (_Float16)t2;      // D-layout -> LDS
            dnx[r] += t2 * wim2x;                     // skip-connection grad
            dny[r] += t2 * wim2y;
        }
    }
    WAVE_LDS_FENCE();

    // ---- backward GEMM: d1 = t2 @ W2 ----
    v16h C0, C1;
    #pragma unroll
    for (int i = 0; i < 16; ++i) {
        const int kb = i + ((i < 8) ? (half ? 8 : 0) : (half ? 16 : 8));
        C0[i] = tp[wv][lm*64 + kb];                   // LDS -> A-layout
        C1[i] = tp[wv][lm*64 + 32 + kb];
    }
    #pragma unroll
    for (int t = 0; t < 4; ++t) {
        v16h B0, B1;
        #pragma unroll
        for (int i = 0; i < 16; ++i) {
            const int jb = i + ((i < 8) ? (half ? 8 : 0) : (half ? 16 : 8));
            B0[i] = w2h[jb*64        + 16*t + lm];    // B[j][k] = W2[j][k]
            B1[i] = w2h[(32 + jb)*64 + 16*t + lm];
        }
        v8f dacc = {};
        dacc = __builtin_amdgcn_wmma_f32_16x16x32_f16(false, C0, false, B0,
                                                      (short)0, dacc, false, false);
        dacc = __builtin_amdgcn_wmma_f32_16x16x32_f16(false, C1, false, B1,
                                                      (short)0, dacc, false, false);
        const int k = 16*t + lm;
        if (k < 60) {
            const float w1x = w1s[k*2], w1y = w1s[k*2+1];
            const float bb = b1s[k];
            #pragma unroll
            for (int r = 0; r < 8; ++r) {
                const int row = r + (half ? 8 : 0);
                const float rx0 = xb[wv][row][0], rx1 = xb[wv][row][1];
                const float a1 = rx0*w1x + rx1*w1y + bb;   // recompute s1 in D-layout
                const float t1 = dacc[r] * sig_f(a1);
                dnx[r] += t1 * w1x;
                dny[r] += t1 * w1y;
            }
        }
    }

    // ---- reduce gradient over the 60 hidden dims (cross-lane) ----
    #pragma unroll
    for (int r = 0; r < 8; ++r) {
        float vx = dnx[r], vy = dny[r];
        vx += __shfl_xor(vx, 1, 16); vy += __shfl_xor(vy, 1, 16);
        vx += __shfl_xor(vx, 2, 16); vy += __shfl_xor(vy, 2, 16);
        vx += __shfl_xor(vx, 4, 16); vy += __shfl_xor(vy, 4, 16);
        vx += __shfl_xor(vx, 8, 16); vy += __shfl_xor(vy, 8, 16);
        dnx[r] = vx; dny[r] = vy;
    }
    if (lm == 0) {
        #pragma unroll
        for (int r = 0; r < 8; ++r) {
            dns[wv][r + (half ? 8 : 0)][0] = dnx[r];
            dns[wv][r + (half ? 8 : 0)][1] = dny[r];
        }
    }
    WAVE_LDS_FENCE();

    // ---- per-sample finale: stability correction + output (lanes 0-15) ----
    if (half == 0) {
        const long long s = base + lm;
        if (s < N) {
            const float x0 = xb[wv][lm][0], x1 = xb[wv][lm][1];
            const float dvx = dns[wv][lm][0] + g3m*cb[1] + 2.f*EPS_C*(x0 - cb[13]);
            const float dvy = dns[wv][lm][1] + g3m*cb[2] + 2.f*EPS_C*(x1 - cb[14]);
            const float f0 = x0*cb[3] + x1*cb[4];
            const float f1 = x0*cb[5] + x1*cb[6];
            const float g0 = x0*cb[7] + x1*cb[8] + cb[11];
            const float g1 = x0*cb[9] + x1*cb[10] + cb[12];
            const float u  = U[s];
            float sc = dvx*f0 + dvy*f1 + ALPHA_C*Vm - __builtin_fabsf(dvx*g0 + dvy*g1);
            sc = (sc > 0.f) ? sc : 0.f;
            const float nrm = dvx*dvx + dvy*dvy;
            const float k   = sc * __builtin_amdgcn_rcpf(nrm);
            out[s*2 + 0] = f0 - dvx*k + g0*u;
            out[s*2 + 1] = f1 - dvy*k + g1*u;
        }
    }
}

// ---------------------------------------------------------------------------
extern "C" void kernel_launch(void* const* d_in, const int* in_sizes, int n_in,
                              void* d_out, int out_size, void* d_ws, size_t ws_size,
                              hipStream_t stream) {
    const float* X     = (const float*)d_in[0];
    const float* U     = (const float*)d_in[1];
    const float* Xref  = (const float*)d_in[2];
    const float* W_fnn = (const float*)d_in[3];
    const float* W_gnn = (const float*)d_in[4];
    const float* b_gnn = (const float*)d_in[5];
    const float* W1    = (const float*)d_in[6];
    const float* b1    = (const float*)d_in[7];
    const float* W2    = (const float*)d_in[8];
    const float* b2    = (const float*)d_in[9];
    const float* W3    = (const float*)d_in[10];
    const float* b3    = (const float*)d_in[11];
    const float* Wim2  = (const float*)d_in[12];
    const float* Wim3  = (const float*)d_in[13];
    float* out = (float*)d_out;
    float* h0  = (float*)d_ws;

    const int N = in_sizes[0] / 2;

    k_h0<<<1, 64, 0, stream>>>(Xref, W1, b1, W2, b2, W3, b3, Wim2, Wim3, h0);

    const int blocks = (N + 127) / 128;
    k_main<<<blocks, 256, 0, stream>>>(X, U, Xref, W_fnn, W_gnn, b_gnn,
                                       W1, b1, W2, b2, W3, b3, Wim2, Wim3,
                                       h0, out, N);
}
